// Sub_Clust_Cos_GLM_10213432230421
// MI455X (gfx1250) — matrix operand order repaired and checked by reading, not compiled, not run on hardware
//
#include <hip/hip_runtime.h>
#include <hip/hip_bf16.h>
#include <math.h>

// ---------------------------------------------------------------------------
// Sub_Clust_Cos_GLM on MI455X (gfx1250, wave32, WMMA).
//
//   K0 k_softmax : C_syn = softmax(raw*1e4, axis=sub) -> d_out, plus pooled
//                  matrix Cw^T[e][s] = C*exp(scale) -> ws.
//   K1 k_kern    : GLM kernels (W @ raised-cosine basis, 200 taps padded to
//                  224 = 7*32) pre-swizzled into the WMMA bf16 A-operand
//                  per-lane layout -> ws.
//   K2 k_pool    : syn[b][s][t] = sum_e S[b][t][e]*Cw^T[e][s]; HBM-bound
//                  (400 MB non-temporal read ~= 17 us @ 23.3 TB/s). Output
//                  stored as bf16, T-REVERSED, so conv B-fragments are
//                  contiguous in LDS (ds_load_b128-able).
//   K3 k_conv    : causal grouped conv, D[16h x 16t] += A[16h x 32tau](kern,
//                  registers) x B[32tau x 16t](syn window, LDS); 14 WMMAs per
//                  tile; fused tanh + exp(W2) readout.
//   K4 k_final   : final[b][t] = sum_s sub_out + V_o.
// ---------------------------------------------------------------------------

#define B_NO   8
#define T_LEN  10000
#define E_NO   1000
#define I_NO   250
#define SUB    8
#define HID    16
#define CB     30
#define NKC    7            // 224 / 32 K-chunks
#define HIST   224          // kernel length padded to 7*32
#define TILE_T 512          // t-chunk per workgroup in k_conv
#define NCHUNK 20           // ceil(10000/512)
#define WNDW   (HIST + TILE_T)   // 736
#define PI_F   3.14159265358979323846f

typedef __attribute__((ext_vector_type(16))) __bf16 v16bf;
typedef __attribute__((ext_vector_type(8)))  float  v8f;

// float -> bf16 round-to-nearest-even via integer math (no __bf16 arithmetic)
__device__ __forceinline__ unsigned short f2bf(float f) {
    union { float f; unsigned int u; } c; c.f = f;
    unsigned int u = c.u + 0x7FFFu + ((c.u >> 16) & 1u);
    return (unsigned short)(u >> 16);
}

// overflow-safe tanh: 1 - 2*rcp(exp(2x)+1); v_rcp_f32 (~1 ulp) instead of the
// multi-instruction IEEE divide sequence.
__device__ __forceinline__ float fast_tanh(float x) {
    float e = __expf(2.f * x);
    return 1.f - 2.f * __builtin_amdgcn_rcpf(e + 1.f);
}

// ---------------------------------------------------------------------------
// K0: column softmax over the sub axis + scaled pooled matrix Cw^T.
// ---------------------------------------------------------------------------
__global__ void k_softmax(const float* __restrict__ Ce_raw,
                          const float* __restrict__ Ci_raw,
                          const float* __restrict__ Escale,
                          const float* __restrict__ Iscale,
                          float* __restrict__ out_Ce,
                          float* __restrict__ out_Ci,
                          float* __restrict__ cwt_e,
                          float* __restrict__ cwt_i) {
    int j = blockIdx.x * blockDim.x + threadIdx.x;
    if (j < E_NO) {
        float r[SUB]; float m = -3.0e38f;
        #pragma unroll
        for (int s = 0; s < SUB; ++s) { r[s] = Ce_raw[s * E_NO + j] * 10000.f; m = fmaxf(m, r[s]); }
        float d = 0.f;
        #pragma unroll
        for (int s = 0; s < SUB; ++s) { r[s] = __expf(r[s] - m); d += r[s]; }
        float inv = 1.f / d, sc = __expf(Escale[j]);
        #pragma unroll
        for (int s = 0; s < SUB; ++s) {
            float p = r[s] * inv;
            out_Ce[s * E_NO + j] = p;
            cwt_e[j * SUB + s] = p * sc;
        }
    } else if (j < E_NO + I_NO) {
        int k = j - E_NO;
        float r[SUB]; float m = -3.0e38f;
        #pragma unroll
        for (int s = 0; s < SUB; ++s) { r[s] = Ci_raw[s * I_NO + k] * 10000.f; m = fmaxf(m, r[s]); }
        float d = 0.f;
        #pragma unroll
        for (int s = 0; s < SUB; ++s) { r[s] = __expf(r[s] - m); d += r[s]; }
        float inv = 1.f / d, sc = __expf(Iscale[k]);
        #pragma unroll
        for (int s = 0; s < SUB; ++s) {
            float p = r[s] * inv;
            out_Ci[s * I_NO + k] = p;
            cwt_i[k * SUB + s] = p * sc;
        }
    }
}

// ---------------------------------------------------------------------------
// K1: GLM kernels pre-swizzled into the WMMA bf16 A-operand (16h x 32tau)
// per-lane layout: lane L (h = L&15, g = L>>4), element j in [0,16):
//   K = j + 16g (contiguous k order chosen here, same as the B fragments; not AMD's table order)
// frag[((s*2+ei)*NKC + kc)*32 + L][j] = kern[h][tau = kc*32 + K]
// ---------------------------------------------------------------------------
__global__ void k_kern(const float* __restrict__ We,
                       const float* __restrict__ Wi,
                       unsigned short* __restrict__ frag) {
    int s = blockIdx.x >> 1, ei = blockIdx.x & 1;
    const float* W = ei ? Wi : We;
    for (int idx = threadIdx.x; idx < NKC * 32 * 16; idx += blockDim.x) {
        int jj = idx & 15;
        int L  = (idx >> 4) & 31;
        int kc = idx >> 9;
        int h  = L & 15, g = L >> 4;
        int K  = jj + 16 * g;
        int tau = kc * 32 + K;
        float v = 0.f;
        if (tau < 200) {
            float raw = 7.5f * logf((float)tau + 1.0f + 1e-7f);
            const float* w = &W[(s * HID + h) * CB];
            #pragma unroll
            for (int cb = 0; cb < CB; ++cb) {
                float x = raw - 0.5f * PI_F * (float)cb;
                float bval = (x >= -PI_F && x <= PI_F) ? (0.5f * __cosf(x) + 0.5f) : 0.f;
                v = fmaf(w[cb], bval, v);
            }
        }
        frag[(size_t)(((s * 2 + ei) * NKC + kc) * 32 + L) * 16 + jj] = f2bf(v);
    }
}

// ---------------------------------------------------------------------------
// K2: synapse pooling (the HBM-bound stage). Wave per (b,t); lanes stride the
// contiguous E axis -> fully coalesced non-temporal reads (S is streamed
// exactly once; keep the 192MB L2 for reused data); Cw^T in LDS as float4
// pairs; wave32 shfl_xor reduction. Output: bf16, T-REVERSED (index T-1-t)
// so the conv stage reads contiguous bf16 runs.
// ---------------------------------------------------------------------------
__global__ __launch_bounds__(256) void k_pool(const float* __restrict__ S_e,
                                              const float* __restrict__ S_i,
                                              const float* __restrict__ cwt_e,
                                              const float* __restrict__ cwt_i,
                                              unsigned short* __restrict__ synrev_e,
                                              unsigned short* __restrict__ synrev_i) {
    __shared__ float lds_e[E_NO * SUB];   // 32 KB
    __shared__ float lds_i[I_NO * SUB];   //  8 KB
    for (int k = threadIdx.x; k < E_NO * SUB; k += blockDim.x) lds_e[k] = cwt_e[k];
    for (int k = threadIdx.x; k < I_NO * SUB; k += blockDim.x) lds_i[k] = cwt_i[k];
    __syncthreads();

    int wave = threadIdx.x >> 5, lane = threadIdx.x & 31;
    int wg = blockIdx.x * 8 + wave;            // 10000 waves total
    for (int p = 0; p < 8; ++p) {              // 8 (b,t) pairs per wave
        int bt = wg * 8 + p;
        int b = bt / T_LEN, t = bt % T_LEN;
        float ae[SUB] = {0.f,0.f,0.f,0.f,0.f,0.f,0.f,0.f};
        float ai[SUB] = {0.f,0.f,0.f,0.f,0.f,0.f,0.f,0.f};
        const float* se = &S_e[(size_t)(b * T_LEN + t) * E_NO];
        const float* si = &S_i[(size_t)(b * T_LEN + t) * I_NO];
        for (int e = lane; e < E_NO; e += 32) {
            float sv = __builtin_nontemporal_load(&se[e]);
            const float4* cp = reinterpret_cast<const float4*>(&lds_e[e * SUB]);
            float4 c0 = cp[0], c1 = cp[1];
            ae[0] = fmaf(sv, c0.x, ae[0]); ae[1] = fmaf(sv, c0.y, ae[1]);
            ae[2] = fmaf(sv, c0.z, ae[2]); ae[3] = fmaf(sv, c0.w, ae[3]);
            ae[4] = fmaf(sv, c1.x, ae[4]); ae[5] = fmaf(sv, c1.y, ae[5]);
            ae[6] = fmaf(sv, c1.z, ae[6]); ae[7] = fmaf(sv, c1.w, ae[7]);
        }
        for (int e = lane; e < I_NO; e += 32) {
            float sv = __builtin_nontemporal_load(&si[e]);
            const float4* cp = reinterpret_cast<const float4*>(&lds_i[e * SUB]);
            float4 c0 = cp[0], c1 = cp[1];
            ai[0] = fmaf(sv, c0.x, ai[0]); ai[1] = fmaf(sv, c0.y, ai[1]);
            ai[2] = fmaf(sv, c0.z, ai[2]); ai[3] = fmaf(sv, c0.w, ai[3]);
            ai[4] = fmaf(sv, c1.x, ai[4]); ai[5] = fmaf(sv, c1.y, ai[5]);
            ai[6] = fmaf(sv, c1.z, ai[6]); ai[7] = fmaf(sv, c1.w, ai[7]);
        }
        #pragma unroll
        for (int s = 0; s < SUB; ++s) {
            float v = ae[s];
            v += __shfl_xor(v, 1);  v += __shfl_xor(v, 2);
            v += __shfl_xor(v, 4);  v += __shfl_xor(v, 8);
            v += __shfl_xor(v, 16);
            ae[s] = v;
            float w = ai[s];
            w += __shfl_xor(w, 1);  w += __shfl_xor(w, 2);
            w += __shfl_xor(w, 4);  w += __shfl_xor(w, 8);
            w += __shfl_xor(w, 16);
            ai[s] = w;
        }
        if (lane == 0) {
            int tr = T_LEN - 1 - t;
            #pragma unroll
            for (int s = 0; s < SUB; ++s) {
                synrev_e[(size_t)(b * SUB + s) * T_LEN + tr] = f2bf(ae[s]);
                synrev_i[(size_t)(b * SUB + s) * T_LEN + tr] = f2bf(ai[s]);
            }
        }
    }
}

// ---------------------------------------------------------------------------
// K3: causal grouped conv via V_WMMA_F32_16X16X32_BF16.
// D[16h x 16t] += A[16h x 32tau](kernel, registers) x B[32tau x 16t].
// LDS holds the reversed bf16 syn window: slot k <-> t = t_base+TILE_T-1-k,
// so the B fragment for lane L (n = L&15, g = L>>4, elt j -> K = 16g+j) is
// 16 CONTIGUOUS ushorts starting at rb = 511 - tile*16 - n + 16g + 32kc
// (lowered by the compiler to ds_load_b128 pairs feeding the WMMAs).
// Epilogue: lane holds h-rows r+8g of column t0+n -> 8-term FMA chain +
// one shfl_xor(16) + coalesced store by lanes 0..15.
// ---------------------------------------------------------------------------
__global__ __launch_bounds__(128) void k_conv(const unsigned short* __restrict__ synrev_e,
                                              const unsigned short* __restrict__ synrev_i,
                                              const unsigned short* __restrict__ frag_raw,
                                              const float* __restrict__ W2,
                                              const float* __restrict__ b1,
                                              float* __restrict__ sub_out) {
    __shared__ unsigned short lre[WNDW];   // 736 bf16
    __shared__ unsigned short lri[WNDW];

    int chunk = blockIdx.x % NCHUNK;
    int s     = (blockIdx.x / NCHUNK) % SUB;
    int b     = blockIdx.x / (NCHUNK * SUB);
    int t_base = chunk * TILE_T;

    // LDS slot k <-> t = t_base + TILE_T - 1 - k <-> reversed global index
    // grev = (T - t_base - TILE_T) + k; zero outside [0, T).
    const unsigned short* sre = &synrev_e[(size_t)(b * SUB + s) * T_LEN];
    const unsigned short* sri = &synrev_i[(size_t)(b * SUB + s) * T_LEN];
    int grev0 = T_LEN - t_base - TILE_T;
    for (int k = threadIdx.x; k < WNDW; k += blockDim.x) {
        int gr = grev0 + k;
        bool ok = (gr >= 0) && (gr < T_LEN);
        lre[k] = ok ? sre[gr] : (unsigned short)0;
        lri[k] = ok ? sri[gr] : (unsigned short)0;
    }
    __syncthreads();

    int wave = threadIdx.x >> 5, lane = threadIdx.x & 31;
    int g = lane >> 4, n = lane & 15;

    // Kernel (A operand) fragments: constant per subunit, in registers.
    const v16bf* frag = reinterpret_cast<const v16bf*>(frag_raw);
    v16bf akE[NKC], akI[NKC];
    #pragma unroll
    for (int kc = 0; kc < NKC; ++kc) {
        akE[kc] = frag[((s * 2 + 0) * NKC + kc) * 32 + lane];
        akI[kc] = frag[((s * 2 + 1) * NKC + kc) * 32 + lane];
    }
    // Per-row epilogue constants: row r of this lane is h = r + 8g.
    float bias[8], w2r[8];
    #pragma unroll
    for (int r = 0; r < 8; ++r) {
        int hh = r + 8 * g;
        bias[r] = b1[s * HID + hh];
        w2r[r]  = __expf(W2[s * HID + hh]);
    }

    int chunk_len = min(TILE_T, T_LEN - t_base);
    for (int tile = wave; tile * 16 < chunk_len; tile += 4) {
        int t0  = t_base + tile * 16;
        int rb0 = TILE_T - 1 - tile * 16 - n + 16 * g;   // + 32*kc per chunk
        v8f acc = {};
        #pragma unroll
        for (int kc = 0; kc < NKC; ++kc) {
            int rb = rb0 + kc * 32;
            union { v16bf v; unsigned short u[16]; } bE, bI;
            #pragma unroll
            for (int jj = 0; jj < 16; ++jj) {
                bE.u[jj] = lre[rb + jj];
                bI.u[jj] = lri[rb + jj];
            }
            acc = __builtin_amdgcn_wmma_f32_16x16x32_bf16(false, akE[kc], false, bE.v,
                                                          (short)0, acc, false, false);
            acc = __builtin_amdgcn_wmma_f32_16x16x32_bf16(false, akI[kc], false, bI.v,
                                                          (short)0, acc, false, false);
        }
        float p = 0.f;
        #pragma unroll
        for (int r = 0; r < 8; ++r)
            p = fmaf(fast_tanh(acc[r] + bias[r]), w2r[r], p);
        p += __shfl_xor(p, 16);               // combine h = 0..7 with h = 8..15
        if (lane < 16) {
            int t = t0 + n;
            sub_out[(size_t)(b * T_LEN + t) * SUB + s] = p;
        }
    }
}

// ---------------------------------------------------------------------------
// K4: final[b][t] = sum_s sub_out[b][t][s] + V_o
// ---------------------------------------------------------------------------
__global__ void k_final(const float* __restrict__ sub_out,
                        const float* __restrict__ V_o,
                        float* __restrict__ out_final) {
    int idx = blockIdx.x * blockDim.x + threadIdx.x;
    if (idx < B_NO * T_LEN) {
        const float4* p = reinterpret_cast<const float4*>(&sub_out[(size_t)idx * SUB]);
        float4 a = p[0], c = p[1];
        out_final[idx] = V_o[0] + a.x + a.y + a.z + a.w + c.x + c.y + c.z + c.w;
    }
}

// ---------------------------------------------------------------------------
extern "C" void kernel_launch(void* const* d_in, const int* in_sizes, int n_in,
                              void* d_out, int out_size, void* d_ws, size_t ws_size,
                              hipStream_t stream) {
    const float* S_e    = (const float*)d_in[0];
    const float* S_i    = (const float*)d_in[1];
    const float* Escale = (const float*)d_in[2];
    const float* Iscale = (const float*)d_in[3];
    const float* We1    = (const float*)d_in[4];
    const float* Wi1    = (const float*)d_in[5];
    const float* W2     = (const float*)d_in[6];
    const float* b1     = (const float*)d_in[7];
    const float* Ce_raw = (const float*)d_in[8];
    const float* Ci_raw = (const float*)d_in[9];
    const float* V_o    = (const float*)d_in[10];

    float* out       = (float*)d_out;
    float* out_final = out;                    //  80000 floats
    float* out_sub   = out + 80000;            // 640000 floats
    float* out_Ce    = out + 720000;           //   8000 floats
    float* out_Ci    = out + 728000;           //   2000 floats

    char* ws = (char*)d_ws;
    float*          cwt_e    = (float*)(ws);                   // 32000 B (pad 32768)
    float*          cwt_i    = (float*)(ws + 32768);           //  8000 B (pad  8192)
    unsigned short* synrev_e = (unsigned short*)(ws + 40960);  // 64*10000*2 = 1,280,000
    unsigned short* synrev_i = (unsigned short*)(ws + 40960 + 1280000);
    unsigned short* fragA    = (unsigned short*)(ws + 40960 + 2 * 1280000); // 114,688 B (32B aligned)

    k_softmax<<<(E_NO + I_NO + 255) / 256, 256, 0, stream>>>(
        Ce_raw, Ci_raw, Escale, Iscale, out_Ce, out_Ci, cwt_e, cwt_i);

    k_kern<<<SUB * 2, 128, 0, stream>>>(We1, Wi1, fragA);

    k_pool<<<1250, 256, 0, stream>>>(S_e, S_i, cwt_e, cwt_i, synrev_e, synrev_i);

    k_conv<<<B_NO * SUB * NCHUNK, 128, 0, stream>>>(
        synrev_e, synrev_i, fragA, W2, b1, out_sub);

    k_final<<<(B_NO * T_LEN + 255) / 256, 256, 0, stream>>>(out_sub, V_o, out_final);
}
